// Qwen3OmniMoeThinkerExperts_13305808683651
// MI455X (gfx1250) — matrix operand order repaired and checked by reading, not compiled, not run on hardware
//
#include <hip/hip_runtime.h>
#include <hip/hip_bf16.h>

#define TT 8192
#define HH 2048
#define II 768
#define EE 32
#define KK 4
#define TK (TT * KK)

typedef __attribute__((ext_vector_type(16))) __bf16 bf16x16;
typedef __attribute__((ext_vector_type(8)))  __bf16 bf16x8;
typedef __attribute__((ext_vector_type(8)))  float  f32x8;
typedef __attribute__((ext_vector_type(4)))  float  f32x4;

__device__ __forceinline__ bf16x16 cvt16(f32x4 a, f32x4 b, f32x4 c, f32x4 d) {
  bf16x16 r;
#pragma unroll
  for (int j = 0; j < 4; ++j) {
    r[j]      = (__bf16)a[j];
    r[4 + j]  = (__bf16)b[j];
    r[8 + j]  = (__bf16)c[j];
    r[12 + j] = (__bf16)d[j];
  }
  return r;
}

__device__ __forceinline__ bf16x16 loadB(const float* p) {  // 16 contiguous fp32 -> bf16
  return cvt16(*(const f32x4*)(p), *(const f32x4*)(p + 4),
               *(const f32x4*)(p + 8), *(const f32x4*)(p + 12));
}

__device__ __forceinline__ float silu(float g) {
  return g * __builtin_amdgcn_rcpf(1.0f + __expf(-g));
}

// ---------------------------------------------------------------- helpers
__global__ __launch_bounds__(256) void k_zero(f32x4* __restrict__ out4, size_t n4,
                                              int* __restrict__ ws_ints) {
  size_t i = (size_t)blockIdx.x * blockDim.x + threadIdx.x;
  size_t stride = (size_t)gridDim.x * blockDim.x;
  f32x4 z = {};
  for (; i < n4; i += stride) out4[i] = z;
  if (blockIdx.x == 0 && threadIdx.x < 192) ws_ints[threadIdx.x] = 0;  // counts+cursors+offsets
}

__global__ __launch_bounds__(256) void k_count(const int* __restrict__ sel,
                                               int* __restrict__ cnts) {
  int i = blockIdx.x * 256 + threadIdx.x;
  if (i < TK) atomicAdd(&cnts[sel[i]], 1);
}

__global__ void k_scan(const int* __restrict__ cnts, int* __restrict__ offs) {
  if (threadIdx.x == 0) {
    int s = 0;
    for (int e = 0; e < EE; ++e) { offs[e] = s; s += cnts[e]; }
  }
}

__global__ __launch_bounds__(256) void k_scatter(const int* __restrict__ sel,
                                                 const float* __restrict__ rw,
                                                 const int* __restrict__ offs,
                                                 int* __restrict__ cur,
                                                 int* __restrict__ tok,
                                                 float* __restrict__ wgt) {
  int i = blockIdx.x * 256 + threadIdx.x;
  if (i < TK) {
    int e = sel[i];
    int pos = offs[e] + atomicAdd(&cur[e], 1);
    tok[pos] = i >> 2;  // token id (i = t*K + k)
    wgt[pos] = rw[i];
  }
}

// ---------------------------------------------------------------- G1: h = silu(x@G^T) * (x@U^T)
// grid: (token-tile x, I/128 y, E z); block: 256 = 8 waves (2 M x 4 N), wave tile 16Mx32N
__global__ __launch_bounds__(256) void k_gate_up(
    const float* __restrict__ X, const float* __restrict__ G,
    const float* __restrict__ U, const int* __restrict__ counts,
    const int* __restrict__ offsets, const int* __restrict__ tok,
    __bf16* __restrict__ h_all) {
  const int e = blockIdx.z;
  const int cnt = counts[e];
  const int mt = blockIdx.x;
  if (mt * 32 >= cnt) return;
  const int base = offsets[e];

  const int w = threadIdx.x >> 5;
  const int lane = threadIdx.x & 31;
  const int wm = w & 1, wn = w >> 1;
  const int half = lane >> 4, lm = lane & 15;
  const int n0 = blockIdx.y * 128 + wn * 16;  // subtiles at n0 and n0+64

  int rowA = mt * 32 + wm * 16 + lm;
  if (rowA >= cnt) rowA = cnt - 1;
  const int tA = tok[base + rowA];

  const float* xrow = X + (size_t)tA * HH + 16 * half;
  const float* g0p = G + ((size_t)e * II + n0) * HH + (size_t)lm * HH + 16 * half;
  const float* g1p = g0p + (size_t)64 * HH;
  const float* u0p = U + ((size_t)e * II + n0) * HH + (size_t)lm * HH + 16 * half;
  const float* u1p = u0p + (size_t)64 * HH;

  f32x8 accg0 = {}, accg1 = {}, accu0 = {}, accu1 = {};
  for (int k0 = 0; k0 < HH; k0 += 32) {
    bf16x16 A = cvt16(*(const f32x4*)(xrow + k0),      *(const f32x4*)(xrow + k0 + 4),
                      *(const f32x4*)(xrow + k0 + 8), *(const f32x4*)(xrow + k0 + 12));
    bf16x16 Bg0 = loadB(g0p + k0);
    bf16x16 Bg1 = loadB(g1p + k0);
    bf16x16 Bu0 = loadB(u0p + k0);
    bf16x16 Bu1 = loadB(u1p + k0);
    accg0 = __builtin_amdgcn_wmma_f32_16x16x32_bf16(false, A, false, Bg0, (short)0, accg0, false, false);
    accg1 = __builtin_amdgcn_wmma_f32_16x16x32_bf16(false, A, false, Bg1, (short)0, accg1, false, false);
    accu0 = __builtin_amdgcn_wmma_f32_16x16x32_bf16(false, A, false, Bu0, (short)0, accu0, false, false);
    accu1 = __builtin_amdgcn_wmma_f32_16x16x32_bf16(false, A, false, Bu1, (short)0, accu1, false, false);
  }

  const int mbase = mt * 32 + wm * 16 + 8 * half;  // C layout: M = vgpr + 8*half
  __bf16* hp = h_all + (size_t)(base + mbase) * II + n0 + lm;
  if (mbase + 8 <= cnt) {  // full tile fast path (no per-element branches)
#pragma unroll
    for (int r = 0; r < 8; ++r) {
      hp[(size_t)r * II]      = (__bf16)(silu(accg0[r]) * accu0[r]);
      hp[(size_t)r * II + 64] = (__bf16)(silu(accg1[r]) * accu1[r]);
    }
  } else {
#pragma unroll
    for (int r = 0; r < 8; ++r) {
      if (mbase + r < cnt) {
        hp[(size_t)r * II]      = (__bf16)(silu(accg0[r]) * accu0[r]);
        hp[(size_t)r * II + 64] = (__bf16)(silu(accg1[r]) * accu1[r]);
      }
    }
  }
}

// ---------------------------------------------------------------- G2: out[t] += w * (h @ D^T)
// grid: (token-tile x, H/128 y, E z); block: 256 = 8 waves, wave tile 16Mx32N
__global__ __launch_bounds__(256) void k_down(
    const __bf16* __restrict__ h_all, const float* __restrict__ D,
    const int* __restrict__ counts, const int* __restrict__ offsets,
    const int* __restrict__ tok, const float* __restrict__ wgt,
    float* __restrict__ out) {
  const int e = blockIdx.z;
  const int cnt = counts[e];
  const int mt = blockIdx.x;
  if (mt * 32 >= cnt) return;
  const int base = offsets[e];

  const int w = threadIdx.x >> 5;
  const int lane = threadIdx.x & 31;
  const int wm = w & 1, wn = w >> 1;
  const int half = lane >> 4, lm = lane & 15;
  const int n0 = blockIdx.y * 128 + wn * 16;  // subtiles at n0 and n0+64

  int rowA = mt * 32 + wm * 16 + lm;
  if (rowA >= cnt) rowA = cnt - 1;

  const __bf16* hrow = h_all + (size_t)(base + rowA) * II + 16 * half;
  const float* d0p = D + ((size_t)e * HH + n0 + lm) * II + 16 * half;
  const float* d1p = d0p + (size_t)64 * II;

  f32x8 acc0 = {}, acc1 = {};
  for (int k0 = 0; k0 < II; k0 += 32) {
    bf16x8 s0 = *(const bf16x8*)(hrow + k0);
    bf16x8 s1 = *(const bf16x8*)(hrow + k0 + 8);
    bf16x16 A;
#pragma unroll
    for (int j = 0; j < 8; ++j) { A[j] = s0[j]; A[8 + j] = s1[j]; }
    bf16x16 B0 = loadB(d0p + k0);
    bf16x16 B1 = loadB(d1p + k0);
    acc0 = __builtin_amdgcn_wmma_f32_16x16x32_bf16(false, A, false, B0, (short)0, acc0, false, false);
    acc1 = __builtin_amdgcn_wmma_f32_16x16x32_bf16(false, A, false, B1, (short)0, acc1, false, false);
  }

  const int mbase = mt * 32 + wm * 16 + 8 * half;
  if (mbase + 8 <= cnt) {
#pragma unroll
    for (int r = 0; r < 8; ++r) {
      int entry = base + mbase + r;
      int t = tok[entry];
      float wv = wgt[entry];
      float* op = out + (size_t)t * HH + n0 + lm;
      unsafeAtomicAdd(op,      wv * acc0[r]);
      unsafeAtomicAdd(op + 64, wv * acc1[r]);
    }
  } else {
#pragma unroll
    for (int r = 0; r < 8; ++r) {
      if (mbase + r < cnt) {
        int entry = base + mbase + r;
        int t = tok[entry];
        float wv = wgt[entry];
        float* op = out + (size_t)t * HH + n0 + lm;
        unsafeAtomicAdd(op,      wv * acc0[r]);
        unsafeAtomicAdd(op + 64, wv * acc1[r]);
      }
    }
  }
}

// ---------------------------------------------------------------- launch
extern "C" void kernel_launch(void* const* d_in, const int* in_sizes, int n_in,
                              void* d_out, int out_size, void* d_ws, size_t ws_size,
                              hipStream_t stream) {
  const float* X   = (const float*)d_in[0];
  const float* RW  = (const float*)d_in[1];
  const int*   SEL = (const int*)d_in[2];
  const float* G   = (const float*)d_in[4];
  const float* U   = (const float*)d_in[5];
  const float* D   = (const float*)d_in[6];
  float* out = (float*)d_out;

  char* ws = (char*)d_ws;
  int*   counts  = (int*)(ws);                            // E ints
  int*   cursors = (int*)(ws + 256);                      // E ints
  int*   offs    = (int*)(ws + 512);                      // E ints
  int*   tok     = (int*)(ws + 1024);                     // TK ints
  float* wgt     = (float*)(ws + 1024 + TK * 4);          // TK floats
  __bf16* h_all  = (__bf16*)(ws + 1024 + TK * 8 + 1024);  // TK*II bf16, 256B aligned

  const size_t n4 = (size_t)TT * HH / 4;
  k_zero<<<4096, 256, 0, stream>>>((f32x4*)out, n4, counts);
  k_count<<<TK / 256, 256, 0, stream>>>(SEL, counts);
  k_scan<<<1, 32, 0, stream>>>(counts, offs);
  k_scatter<<<TK / 256, 256, 0, stream>>>(SEL, RW, offs, cursors, tok, wgt);

  dim3 g1(1024, II / 128, EE);  // token tiles fastest -> expert weights hot in L2
  k_gate_up<<<g1, 256, 0, stream>>>(X, G, U, counts, offs, tok, h_all);

  dim3 g2(1024, HH / 128, EE);
  k_down<<<g2, 256, 0, stream>>>(h_all, D, counts, offs, tok, wgt, out);
}
